// LocalLinear_57904749084763
// MI455X (gfx1250) — matrix operand order repaired and checked by reading, not compiled, not run on hardware
//
#include <hip/hip_runtime.h>

// ---------------------------------------------------------------------------
// LocalLinear on MI455X (gfx1250, wave32, WMMA):
//   out[b, f, l] = sum_k x[b, 64*f + k] * W[f, l, k] + bias[f, l]
// 127 independent (2048x128)x(128x128)^T GEMMs.
//
// Precision: fp32 inputs split into bf16 hi/lo; D = Ahi*Bhi + Ahi*Blo + Alo*Bhi
// accumulated in fp32 via v_wmma_f32_16x16x32_bf16 (~16-bit operand mantissa,
// fp32 accumulate — fp32-emulation a la 3xBF16). 8x denser per-instruction
// FLOPs than V_WMMA_F32_16X16X4_F32 keeps us near the 23.3 TB/s roofline.
// ---------------------------------------------------------------------------

#define IN_FEAT   8192
#define KSZ       128     // kernel width (= GEMM K)
#define STRIDE_W  64
#define FOLDS     127
#define LF        128     // local_features (= GEMM N)
#define BATCH     2048
#define OUT_W     (FOLDS * LF)   // 16256

#define MBLK      128     // batch rows per workgroup
#define PITCH     136     // LDS row pitch in halfs (128 + 8 pad)

typedef __attribute__((ext_vector_type(16))) __bf16         v16bf;
typedef __attribute__((ext_vector_type(8)))  float          v8f;
typedef __attribute__((ext_vector_type(8)))  unsigned short u16x8;
typedef __attribute__((ext_vector_type(4)))  unsigned short u16x4;

union FragU { u16x8 h[2]; v16bf v; };

struct HiLo { unsigned short hi, lo; };

// Split one float into bf16 hi (RNE) and bf16 lo (residual).
__device__ __forceinline__ HiLo split_bf16(float a) {
    __bf16 h = (__bf16)a;
    float  r = a - (float)h;
    __bf16 l = (__bf16)r;
    HiLo out;
    out.hi = __builtin_bit_cast(unsigned short, h);
    out.lo = __builtin_bit_cast(unsigned short, l);
    return out;
}

__global__ __launch_bounds__(256)
void local_linear_wmma_bf16split(const float* __restrict__ x,
                                 const float* __restrict__ W,
                                 const float* __restrict__ bias,
                                 float* __restrict__ out)
{
    // bf16 hi/lo staging: A = x window tile (row-major, K contiguous),
    //                     B = W[f] rows (col-major for WMMA = (l,k) as stored).
    __shared__ unsigned short sAhi[MBLK * PITCH];
    __shared__ unsigned short sAlo[MBLK * PITCH];
    __shared__ unsigned short sBhi[LF   * PITCH];
    __shared__ unsigned short sBlo[LF   * PITCH];

    const int f    = blockIdx.x;          // fold index
    const int m0   = blockIdx.y * MBLK;   // batch-row base
    const int t    = threadIdx.x;
    const int lane = t & 31;
    const int wave = t >> 5;

    // ------------------- stage: global -> split -> LDS -------------------
    // Per iteration: each wave owns one row; lanes cover 128 floats (512 B,
    // fully coalesced float4 loads). 16 iterations cover all 128 rows.
    const float* xs = x + (size_t)m0 * IN_FEAT + (size_t)STRIDE_W * f;
    const float* ws = W + (size_t)f * LF * KSZ;
    #pragma unroll
    for (int it = 0; it < 16; ++it) {
        const int row = wave + it * 8;
        const int c4  = lane * 4;
        float4 vx = *(const float4*)(xs + (size_t)row * IN_FEAT + c4);
        float4 vw = *(const float4*)(ws + row * KSZ + c4);

        const HiLo x0 = split_bf16(vx.x), x1 = split_bf16(vx.y),
                   x2 = split_bf16(vx.z), x3 = split_bf16(vx.w);
        const HiLo w0 = split_bf16(vw.x), w1 = split_bf16(vw.y),
                   w2 = split_bf16(vw.z), w3 = split_bf16(vw.w);

        const u16x4 xh = {x0.hi, x1.hi, x2.hi, x3.hi};
        const u16x4 xl = {x0.lo, x1.lo, x2.lo, x3.lo};
        const u16x4 wh = {w0.hi, w1.hi, w2.hi, w3.hi};
        const u16x4 wl = {w0.lo, w1.lo, w2.lo, w3.lo};

        const int o = row * PITCH + c4;     // 8B-aligned, conflict-free stores
        *(u16x4*)&sAhi[o] = xh;  *(u16x4*)&sAlo[o] = xl;
        *(u16x4*)&sBhi[o] = wh;  *(u16x4*)&sBlo[o] = wl;
    }
    __syncthreads();

    // ------------------------- compute: WMMA ------------------------------
    // Wave grid 4(M) x 2(N): each wave -> 32 rows x 64 cols = 2x4 C tiles.
    const int wm    = wave >> 1;
    const int wn    = wave & 1;
    const int mBase = wm * 32;
    const int nBase = wn * 64;
    const int lhalf = lane >> 4;   // 0 | 1
    const int lmod  = lane & 15;

    v8f acc[2][4] = {};

    #pragma unroll
    for (int k0 = 0; k0 < KSZ; k0 += 32) {
        // A fragments (16x32, contiguous k order, same as B):
        // lane<16 : row=lane,    K = k0 + 0..15 (contiguous)
        // lane>=16: row=lane-16, K = k0 + 16..31
        v16bf ahi[2], alo[2];
        #pragma unroll
        for (int mt = 0; mt < 2; ++mt) {
            const int row = mBase + mt * 16 + lmod;
            const int ka  = k0 + lhalf * 16;
            FragU u;
            u.h[0] = *(const u16x8*)&sAhi[row * PITCH + ka];
            u.h[1] = *(const u16x8*)&sAhi[row * PITCH + ka + 8];
            ahi[mt] = u.v;
            u.h[0] = *(const u16x8*)&sAlo[row * PITCH + ka];
            u.h[1] = *(const u16x8*)&sAlo[row * PITCH + ka + 8];
            alo[mt] = u.v;
        }
        // B fragments (32x16, contiguous k order, same as A):
        // lane<16 : col=lane,    K = k0 + 0..15 (contiguous)
        // lane>=16: col=lane-16, K = k0 + 16..31
        v16bf bhi[4], blo[4];
        #pragma unroll
        for (int nt = 0; nt < 4; ++nt) {
            const int col = nBase + nt * 16 + lmod;
            const int kb  = k0 + lhalf * 16;
            FragU u;
            u.h[0] = *(const u16x8*)&sBhi[col * PITCH + kb];
            u.h[1] = *(const u16x8*)&sBhi[col * PITCH + kb + 8];
            bhi[nt] = u.v;
            u.h[0] = *(const u16x8*)&sBlo[col * PITCH + kb];
            u.h[1] = *(const u16x8*)&sBlo[col * PITCH + kb + 8];
            blo[nt] = u.v;
        }
        // 24 WMMAs per K-step: hi*hi + hi*lo + lo*hi, fp32 accumulate.
        #pragma unroll
        for (int mt = 0; mt < 2; ++mt) {
            #pragma unroll
            for (int nt = 0; nt < 4; ++nt) {
                acc[mt][nt] = __builtin_amdgcn_wmma_f32_16x16x32_bf16(
                    false, alo[mt], false, bhi[nt], (short)0, acc[mt][nt], false, false);
                acc[mt][nt] = __builtin_amdgcn_wmma_f32_16x16x32_bf16(
                    false, ahi[mt], false, blo[nt], (short)0, acc[mt][nt], false, false);
                acc[mt][nt] = __builtin_amdgcn_wmma_f32_16x16x32_bf16(
                    false, ahi[mt], false, bhi[nt], (short)0, acc[mt][nt], false, false);
            }
        }
    }

    // ------------------------- epilogue: bias + store ---------------------
    // C/D layout: lane L, VGPR v -> (M = v + 8*(L>=16), N = L%16).
    float* op = out + (size_t)m0 * OUT_W + (size_t)f * LF;
    #pragma unroll
    for (int nt = 0; nt < 4; ++nt) {
        const int col = nBase + nt * 16 + lmod;
        const float bv = bias[f * LF + col];
        #pragma unroll
        for (int mt = 0; mt < 2; ++mt) {
            const int rbase = mBase + mt * 16 + lhalf * 8;
            #pragma unroll
            for (int v = 0; v < 8; ++v) {
                op[(size_t)(rbase + v) * OUT_W + col] = acc[mt][nt][v] + bv;
            }
        }
    }
}

extern "C" void kernel_launch(void* const* d_in, const int* in_sizes, int n_in,
                              void* d_out, int out_size, void* d_ws, size_t ws_size,
                              hipStream_t stream) {
    (void)in_sizes; (void)n_in; (void)out_size; (void)d_ws; (void)ws_size;
    const float* x = (const float*)d_in[0];
    const float* W = (const float*)d_in[1];
    const float* b = (const float*)d_in[2];
    float* out     = (float*)d_out;

    dim3 grid(FOLDS, BATCH / MBLK);   // 127 x 16 workgroups
    local_linear_wmma_bf16split<<<grid, 256, 0, stream>>>(x, W, b, out);
}
